// FusedModConv_49503793054288
// MI455X (gfx1250) — matrix operand order repaired and checked by reading, not compiled, not run on hardware
//
#include <hip/hip_runtime.h>
#include <hip/hip_bf16.h>

// ---------------------------------------------------------------------------
// StyleGAN2 modulated upsample-conv, fused for gfx1250 (MI455X).
//   convT(x2) + 4x4 blur  ==  4 per-sample 3x3 convs (one per output phase)
//   with blur-folded weights G.  Implicit GEMM on v_wmma_f32_16x16x32_bf16.
//   A = weights (M=oc, K=ic), B = pixels (K=ic, N=pixel col)  so that the
//   D-fragment lanes index pixel columns -> both px phases held in registers
//   -> contiguous global_store_b64 epilogue (output = 537MB dominates roofline).
// ---------------------------------------------------------------------------

typedef __attribute__((ext_vector_type(16))) __bf16     v16bf;
typedef __attribute__((ext_vector_type(8)))  float      v8f;
typedef __attribute__((ext_vector_type(2)))  float      f32x2;
typedef __attribute__((ext_vector_type(4)))  unsigned   u32x4;

#define B_SZ   8
#define IC     64
#define OC     64
#define H_IN   256
#define WDIM   512

#define TSX    16      // coarse tile width  (pixel columns, = WMMA N)
#define TSY    8       // coarse tile height (one row per wave)
#define RGX    18      // staged region width  (halo 1 each side)
#define RGY    10      // staged region height (halo 1 each side)
#define ICPAD  72      // bf16 elems per (y,x): 64 ic + pad -> 144B row pitch

__device__ __forceinline__ unsigned short f2bf(float f) {
  unsigned u = __builtin_bit_cast(unsigned, f);
  u += 0x7FFFu + ((u >> 16) & 1u);              // round-to-nearest-even
  return (unsigned short)(u >> 16);
}

struct FragBits { u32x4 lo, hi; };
union FragCast { FragBits p; v16bf v; };
__device__ __forceinline__ v16bf make_frag(u32x4 lo, u32x4 hi) {
  FragCast f; f.p.lo = lo; f.p.hi = hi; return f.v;
}

// ---------------------------------------------------------------------------
// Kernel 1: style GEMV + demod + fold 4x4 blur into the modulated 3x3 kernel
// -> per-phase 3x3 coarse kernels, bf16, into d_ws.
// Layout: weff[b][phase(py*2+px)][tap((dy+1)*3+(dx+1))][oc][ic]
// ---------------------------------------------------------------------------
__global__ __launch_bounds__(256)
void modconv_prepare(const float* __restrict__ w,
                     const float* __restrict__ w_conv,
                     const float* __restrict__ w_dense,
                     const float* __restrict__ b_mod,
                     unsigned short* __restrict__ weff) {
  const int b   = blockIdx.x;
  const int tid = threadIdx.x;
  __shared__ float style[IC];
  __shared__ float dvec[OC];
  const float coef_dense = 0.04419417382415922f;  // 1/sqrt(512)
  const float coef_conv  = 1.0f / 24.0f;          // 1/sqrt(3*3*64)

  if (tid < IC) {
    float s = 0.f;
    for (int j = 0; j < WDIM; ++j)
      s += w[b * WDIM + j] * w_dense[j * IC + tid];
    style[tid] = s * coef_dense + b_mod[tid] + 1.0f;
  }
  __syncthreads();

  if (tid < OC) {
    float ss = 0.f;
    for (int a = 0; a < 9; ++a)
      for (int ic = 0; ic < IC; ++ic) {
        float v = coef_conv * w_conv[(a * IC + ic) * OC + tid] * style[ic];
        ss += v * v;
      }
    dvec[tid] = rsqrtf(ss + 1e-8f);
  }
  __syncthreads();

  const float b1[4] = {1.f, 3.f, 3.f, 1.f};       // blur taps (symmetric)
  for (int p = tid; p < OC * IC; p += blockDim.x) {
    const int ic = p & (IC - 1);
    const int oc = p >> 6;
    float w9[9];
    #pragma unroll
    for (int a = 0; a < 9; ++a)
      w9[a] = coef_conv * w_conv[(a * IC + ic) * OC + oc] * style[ic] * dvec[oc];
    // composite fine-grid kernel G[6][6] = w9 (*) blur  (blur scaled 4/64)
    float G[6][6];
    #pragma unroll
    for (int sy = 0; sy < 6; ++sy)
      #pragma unroll
      for (int sx = 0; sx < 6; ++sx) {
        float acc = 0.f;
        #pragma unroll
        for (int ay = 0; ay < 3; ++ay) {
          const int cy = sy - ay;
          if (cy < 0 || cy > 3) continue;
          #pragma unroll
          for (int ax = 0; ax < 3; ++ax) {
            const int cx = sx - ax;
            if (cx < 0 || cx > 3) continue;
            acc += w9[ay * 3 + ax] * (b1[cy] * b1[cx] * (1.0f / 16.0f));
          }
        }
        G[sy][sx] = acc;
      }
    // out(2m+py, 2n+px) accumulates x(m+dy, n+dx) * G[2dy+3-py][2dx+3-px]
    #pragma unroll
    for (int py = 0; py < 2; ++py)
      #pragma unroll
      for (int px = 0; px < 2; ++px)
        #pragma unroll
        for (int ty = 0; ty < 3; ++ty)
          #pragma unroll
          for (int tx = 0; tx < 3; ++tx) {
            const int sy = 2 * ty + 1 - py;
            const int sx = 2 * tx + 1 - px;
            const size_t idx =
                ((((size_t)(b * 4 + py * 2 + px) * 9 + ty * 3 + tx) * OC + oc) * IC + ic);
            weff[idx] = f2bf(G[sy][sx]);
          }
  }
}

// ---------------------------------------------------------------------------
// Kernel 2: per (b, py, 8x16 coarse tile), BOTH px phases.
//   Per wave: 1 pixel row (N) x 4 oc subtiles (M) x 2 px  = 8 accumulators.
//   K = 9 taps x 64 ic in chunks of 32  -> 18 chunks x 8 = 144 WMMA / wave.
// ---------------------------------------------------------------------------
__global__ __launch_bounds__(256)
void modconv_main(const float* __restrict__ x,
                  const unsigned short* __restrict__ weff,
                  float* __restrict__ out) {
  __shared__ unsigned short xt[RGY * RGX * ICPAD];   // 25.9 KB bf16 [y][x][ic]

  const int tid = threadIdx.x;
  const int z   = blockIdx.z;
  const int b   = z >> 1;
  const int py  = z & 1;
  const int ty0 = blockIdx.y * TSY;
  const int tx0 = blockIdx.x * TSX;

  // ---- stage x tile (fp32 -> bf16), zero-padded halo -----------------------
  const float* xb = x + (size_t)b * IC * H_IN * H_IN;
  for (int idx = tid; idx < IC * RGY * RGX; idx += 256) {
    const int ic = idx / (RGY * RGX);
    const int p  = idx - ic * (RGY * RGX);
    const int yy = p / RGX;
    const int xx = p - yy * RGX;
    const int gy = ty0 - 1 + yy;
    const int gx = tx0 - 1 + xx;
    float v = 0.f;
    if ((unsigned)gy < H_IN && (unsigned)gx < H_IN)
      v = xb[((size_t)ic * H_IN + gy) * H_IN + gx];
    xt[(yy * RGX + xx) * ICPAD + ic] = f2bf(v);
  }
  __syncthreads();

  const int lane = tid & 31;
  const int wid  = tid >> 5;         // this wave's pixel row within the tile
  const int half = lane >> 4;        // K-half select per WMMA A/B layouts
  const int nl   = lane & 15;        // A: oc row within subtile; B/D: pixel col

  // weights for the two px phases of this py
  const unsigned short* wb0 = weff + (size_t)(b * 4 + py * 2 + 0) * 9 * OC * IC;
  const unsigned short* wb1 = weff + (size_t)(b * 4 + py * 2 + 1) * 9 * OC * IC;

  v8f acc0[4] = {};                  // px = 0, oc subtiles 0..3
  v8f acc1[4] = {};                  // px = 1

  #pragma unroll
  for (int tap = 0; tap < 9; ++tap) {
    const int dy = tap / 3;          // 0..2 == offset -1..1 (halo baked in)
    const int dx = tap % 3;
    if (tap < 8) {
      __builtin_prefetch(wb0 + (size_t)(tap + 1) * OC * IC, 0, 3);
      __builtin_prefetch(wb1 + (size_t)(tap + 1) * OC * IC, 0, 3);
    }
    #pragma unroll
    for (int icb = 0; icb < 2; ++icb) {
      // B fragment (x pixels): lane = pixel col, half selects K octets (8h+e, 16+8h+e).
      const unsigned short* bp =
          &xt[((wid + dy) * RGX + (nl + dx)) * ICPAD + icb * 32 + half * 8];
      const v16bf Bx = make_frag(*(const u32x4*)bp, *(const u32x4*)(bp + 16));
      #pragma unroll
      for (int n = 0; n < 4; ++n) {
        // A fragments (weights): lane = oc row, half selects K octets
        // (elems 0-7 -> ic = 8h+e, elems 8-15 -> ic = 16+8h+e within chunk).
        const size_t wi = ((size_t)tap * OC + n * 16 + nl) * IC + icb * 32 + half * 8;
        const v16bf A0 = make_frag(*(const u32x4*)(wb0 + wi),
                                   *(const u32x4*)(wb0 + wi + 16));
        const v16bf A1 = make_frag(*(const u32x4*)(wb1 + wi),
                                   *(const u32x4*)(wb1 + wi + 16));
        acc0[n] = __builtin_amdgcn_wmma_f32_16x16x32_bf16(
            false, A0, false, Bx, (short)0, acc0[n], false, false);
        acc1[n] = __builtin_amdgcn_wmma_f32_16x16x32_bf16(
            false, A1, false, Bx, (short)0, acc1[n], false, false);
      }
    }
  }

  // ---- epilogue: lane = pixel col, vgpr v -> oc = n*16 + v + 8*half --------
  // px pair in registers -> b64 stores; 16 lanes cover a contiguous 128B run.
  const int gy = ty0 + wid;
  const int fy = 2 * gy + py;
  const int fx = 2 * (tx0 + nl);
  float* ob = out + ((size_t)b * OC * 512 + fy) * 512 + fx;
  #pragma unroll
  for (int n = 0; n < 4; ++n) {
    #pragma unroll
    for (int v = 0; v < 8; ++v) {
      const int oc = n * 16 + v + 8 * half;
      f32x2 pr;
      pr.x = acc0[n][v];
      pr.y = acc1[n][v];
      *(f32x2*)(ob + (size_t)oc * 512 * 512) = pr;
    }
  }
}

// ---------------------------------------------------------------------------
extern "C" void kernel_launch(void* const* d_in, const int* in_sizes, int n_in,
                              void* d_out, int out_size, void* d_ws, size_t ws_size,
                              hipStream_t stream) {
  (void)in_sizes; (void)n_in; (void)out_size; (void)ws_size;
  const float* x       = (const float*)d_in[0];
  const float* w       = (const float*)d_in[1];
  const float* w_conv  = (const float*)d_in[2];
  const float* w_dense = (const float*)d_in[3];
  const float* b_mod   = (const float*)d_in[4];
  unsigned short* weff = (unsigned short*)d_ws;   // 8*4*9*64*64 bf16 = 2.3 MB

  modconv_prepare<<<dim3(B_SZ), dim3(256), 0, stream>>>(w, w_conv, w_dense, b_mod, weff);
  modconv_main<<<dim3(H_IN / TSX, H_IN / TSY, B_SZ * 2), dim3(256), 0, stream>>>(
      x, weff, (float*)d_out);
}
